// InterlacedSparseSelfAttention_31275951850306
// MI455X (gfx1250) — matrix operand order repaired and checked by reading, not compiled, not run on hardware
//
#include <hip/hip_runtime.h>
#include <hip/hip_bf16.h>

// MI455X (gfx1250) interlaced sparse self-attention.
// - One workgroup per attention problem; all operands staged in LDS as bf16.
// - Every GEMM via v_wmma_f32_16x16x32_bf16 (fp32 accumulate).
// - Weights DMA'd into LDS with the Tensor Data Mover (tensor_load_to_lds),
//   completion via s_wait_tensorcnt.
// - Pass 2 runs in place on the pass-1 output (disjoint per-block tiles).

typedef __attribute__((ext_vector_type(16))) __bf16 v16bf;
typedef __attribute__((ext_vector_type(8)))  __bf16 bf16x8;
typedef __attribute__((ext_vector_type(8)))  float  v8f;
typedef __attribute__((ext_vector_type(4)))  unsigned int u32x4;
typedef __attribute__((ext_vector_type(8)))  unsigned int u32x8;

static __device__ __forceinline__ v16bf mk16(bf16x8 lo, bf16x8 hi) {
  union U { v16bf v; bf16x8 h[2]; } u;
  u.h[0] = lo; u.h[1] = hi;
  return u.v;
}

static __device__ __forceinline__ v8f wmma_bf16(v16bf a, v16bf b, v8f c) {
  // (neg_a, A, neg_b, B, c_mod, C, reuse_a, reuse_b)
  return __builtin_amdgcn_wmma_f32_16x16x32_bf16(false, a, false, b, (short)0, c,
                                                 false, false);
}

// ---- Tensor Data Mover: 1-D fp32 copy global -> LDS (D# per ISA ch.8) ----
// Group0: [1:0]=count=1, lds_addr, global_addr[56:0], type=2 ("image")
// Group1: data_size=4B (code 2), tensor_dim0=n, tile_dim0=n (1-D tile)
static __device__ __forceinline__ void tdm_load_1d_f32(unsigned lds_addr,
                                                       const void* gptr,
                                                       unsigned n /* elements, <65536 */) {
  unsigned long long ga = (unsigned long long)(uintptr_t)gptr;
  u32x4 g0;
  g0[0] = 1u;                                            // count=1, user mode
  g0[1] = lds_addr;                                      // LDS byte address
  g0[2] = (unsigned)ga;                                  // global_addr[31:0]
  g0[3] = ((unsigned)(ga >> 32) & 0x01FFFFFFu) | (2u << 30);  // [56:32] | type=2
  u32x8 g1;
  g1[0] = 2u << 16;                                      // data_size = 4 bytes
  g1[1] = (n & 0xFFFFu) << 16;                           // tensor_dim0[15:0]
  g1[2] = (n >> 16) & 0xFFFFu;                           // tensor_dim0[31:16]
  g1[3] = (n & 0xFFFFu) << 16;                           // tile_dim0
  g1[4] = 0u;                                            // tile_dim1/2 unused
  g1[5] = n;                                             // tensor_dim0_stride[31:0]
  g1[6] = 0u;
  g1[7] = 0u;
  // 2 SGPR-group form (tensors up to 2D); VADDR2/3 NULL.
  asm volatile("tensor_load_to_lds %0, %1" :: "s"(g0), "s"(g1) : "memory");
}

// LDS byte offsets
static constexpr int OFF_WQKV  = 0;                       // bf16 [192][72]
static constexpr int OFF_WPROJ = 192 * 72 * 2;            // bf16 [64][72]
static constexpr int OFF_BIAS  = OFF_WPROJ + 64 * 72 * 2; // f32 [64]
static constexpr int OFF_X     = OFF_BIAS + 64 * 4;       // bf16 [N][72]; also fp32 weight staging
static constexpr int STAGE_BYTES = (12288 + 4096) * 4;    // 64 KB fp32 weight staging

static constexpr size_t smem_bytes(int N) {
  size_t base = (size_t)OFF_X
              + (size_t)N * 72 * 2            // X (later O)
              + (size_t)N * 200 * 2           // QKV
              + (size_t)64 * (N + 8) * 2      // V^T
              + (size_t)8 * 16 * (N + 8) * 2; // per-wave P
  size_t stage_end = (size_t)OFF_X + STAGE_BYTES;
  return base > stage_end ? base : stage_end;
}

template <int NTOK, int PASS>
__global__ __launch_bounds__(256, 1)
void isa_attn_kernel(const float* __restrict__ xin,
                     const float* __restrict__ wqkv,
                     const float* __restrict__ wproj,
                     const float* __restrict__ bproj,
                     float* __restrict__ xout) {
  constexpr int RT = NTOK / 16;  // 16-row tiles over tokens
  constexpr int NC = NTOK / 32;  // K=32 chunks over tokens (for P@V)
  constexpr int NP = NTOK + 8;   // padded token stride

  extern __shared__ char smem_raw[];
  __bf16* sWqkv  = (__bf16*)(smem_raw + OFF_WQKV);   // [192][72]
  __bf16* sWproj = (__bf16*)(smem_raw + OFF_WPROJ);  // [64][72]
  float*  sBias  = (float*)(smem_raw + OFF_BIAS);    // [64]
  __bf16* sX     = (__bf16*)(smem_raw + OFF_X);      // [NTOK][72]  X, reused as O
  __bf16* sQKV   = sX + NTOK * 72;                   // [NTOK][200] Q|K|V
  __bf16* sVt    = sQKV + NTOK * 200;                // [64][NP]    V^T (h*8+d, token)
  __bf16* sPall  = sVt + 64 * NP;                    // [8][16][NP] per-wave probs
  float*  stage  = (float*)(smem_raw + OFF_X);       // fp32 weight staging (TDM dest)

  const int tid  = threadIdx.x;
  const int lane = tid & 31;
  const int wv   = tid >> 5;
  const int half = lane >> 4;
  const int l16  = lane & 15;

  int b = 0, ph = 0, pw = 0, qh = 0, qw = 0;
  if (PASS == 1) { b = blockIdx.x >> 6; ph = (blockIdx.x >> 3) & 7; pw = blockIdx.x & 7; }
  else           { b = blockIdx.x >> 8; qh = (blockIdx.x >> 4) & 15; qw = blockIdx.x & 15; }
  const int bbase = b * 64 * 16384;

  // ---- Phase 0a: TDM DMA of the fp32 weights into LDS staging (wave 0) ----
  if (wv == 0) {
    unsigned lds0 = __builtin_amdgcn_groupstaticsize() + (unsigned)OFF_X;
    tdm_load_1d_f32(lds0, wqkv, 192 * 64);
    tdm_load_1d_f32(lds0 + 12288u * 4u, wproj, 64 * 64);
    __builtin_amdgcn_s_wait_tensorcnt(0);
  }
  if (tid < 64) sBias[tid] = bproj[tid];
  __syncthreads();

  // ---- Phase 0b: convert staged weights fp32 -> bf16 (padded rows) ----
  for (int i = tid; i < 192 * 64; i += 256)
    sWqkv[(i >> 6) * 72 + (i & 63)] = (__bf16)stage[i];
  for (int i = tid; i < 64 * 64; i += 256)
    sWproj[(i >> 6) * 72 + (i & 63)] = (__bf16)stage[12288 + i];
  __syncthreads();

  // ---- Phase 1: gather X tokens -> LDS bf16 [token][channel] (overwrites staging)
  for (int i = tid; i < NTOK * 64; i += 256) {
    int ch = i / NTOK, t = i % NTOK;
    int h, w;
    if (PASS == 1) { h = (t >> 4) * 8 + ph; w = (t & 15) * 8 + pw; }
    else           { h = qh * 8 + (t >> 3); w = qw * 8 + (t & 7); }
    sX[t * 72 + ch] = (__bf16)xin[bbase + ch * 16384 + h * 128 + w];
  }
  __syncthreads();

  // ---- Phase 2: QKV GEMM  [NTOK,64] @ [64,192] ----
  // A: lane = row, K split {0-7,16-23}/{8-15,24-31} per half (ISA A-layout)
  // B: lane = column, K split {0-7,16-23}/{8-15,24-31} per half (same as A)
  for (int idx = wv; idx < RT * 12; idx += 8) {
    int rt = idx / 12, ct = idx % 12;
    v8f acc = {};
#pragma unroll
    for (int c0 = 0; c0 < 64; c0 += 32) {
      const __bf16* xr = &sX[(rt * 16 + l16) * 72 + c0 + half * 8];
      v16bf a = mk16(*(const bf16x8*)xr, *(const bf16x8*)(xr + 16));
      const __bf16* wr = &sWqkv[(ct * 16 + l16) * 72 + c0 + half * 8];
      v16bf bb = mk16(*(const bf16x8*)wr, *(const bf16x8*)(wr + 16));
      acc = wmma_bf16(a, bb, acc);
    }
#pragma unroll
    for (int v = 0; v < 8; v++)
      sQKV[(rt * 16 + v + half * 8) * 200 + ct * 16 + l16] = (__bf16)acc[v];
  }
  __syncthreads();

  // ---- Phase 3: transpose V -> sVt[h*8+d][token] ----
  for (int i = tid; i < 64 * NTOK; i += 256) {
    int hd = i / NTOK, t = i % NTOK;
    sVt[hd * NP + t] = sQKV[t * 200 + 128 + hd];
  }
  __syncthreads();

  // ---- Phase 4: attention, one head per wave ----
  {
    const int h = wv;
    __bf16* sP = sPall + wv * 16 * NP;
    for (int qt = 0; qt < RT; qt++) {
      // Q fragment: 16 rows x (d=8 zero-padded to K=32)
      v16bf aq = {};
      if (half == 0) {
        bf16x8 qv = *(const bf16x8*)&sQKV[(qt * 16 + l16) * 200 + h * 8];
        bf16x8 z = {};
        aq = mk16(qv, z);
      }
      v8f S[RT];
#pragma unroll
      for (int kt = 0; kt < RT; kt++) {
        // B = K^T: lane = key token, K(=qdim) contiguous: half0 holds qdim 0..15
        // (0..7 valid), half1 holds qdim 16..31 (all pad zeros).
        v16bf bk = {};
        if (half == 0) {
          bf16x8 kv = *(const bf16x8*)&sQKV[(kt * 16 + l16) * 200 + 64 + h * 8];
          bf16x8 z = {};
          bk = mk16(kv, z);
        }
        v8f zero = {};
        S[kt] = wmma_bf16(aq, bk, zero);
      }
      // row-wise softmax (rows live across a 16-lane half per C-layout)
      const float scale = 0.35355339059327379f;  // 1/sqrt(8)
      float mrow[8], ssum[8];
#pragma unroll
      for (int v = 0; v < 8; v++) mrow[v] = -3.0e38f;
#pragma unroll
      for (int kt = 0; kt < RT; kt++)
#pragma unroll
        for (int v = 0; v < 8; v++) {
          S[kt][v] *= scale;
          mrow[v] = fmaxf(mrow[v], S[kt][v]);
        }
#pragma unroll
      for (int off = 8; off >= 1; off >>= 1)
#pragma unroll
        for (int v = 0; v < 8; v++)
          mrow[v] = fmaxf(mrow[v], __shfl_xor(mrow[v], off));
#pragma unroll
      for (int v = 0; v < 8; v++) ssum[v] = 0.0f;
#pragma unroll
      for (int kt = 0; kt < RT; kt++)
#pragma unroll
        for (int v = 0; v < 8; v++) {
          float e = __expf(S[kt][v] - mrow[v]);
          S[kt][v] = e;
          ssum[v] += e;
        }
#pragma unroll
      for (int off = 8; off >= 1; off >>= 1)
#pragma unroll
        for (int v = 0; v < 8; v++)
          ssum[v] += __shfl_xor(ssum[v], off);
      float sinv[8];
#pragma unroll
      for (int v = 0; v < 8; v++) sinv[v] = 1.0f / ssum[v];
      // normalized probabilities -> wave-private LDS (bf16)
#pragma unroll
      for (int kt = 0; kt < RT; kt++)
#pragma unroll
        for (int v = 0; v < 8; v++)
          sP[(v + half * 8) * NP + kt * 16 + l16] = (__bf16)(S[kt][v] * sinv[v]);

      // O = P @ V  (K = tokens in chunks of 32; head dim pads the N side)
      v8f o = {};
#pragma unroll
      for (int c = 0; c < NC; c++) {
        // A = P: lane = row, K split per half (A-layout)
        const __bf16* pr = &sP[l16 * NP + c * 32 + half * 8];
        v16bf ap = mk16(*(const bf16x8*)pr, *(const bf16x8*)(pr + 16));
        // B = V: lane = dim column (<8 valid), K split per half (same as A)
        v16bf bv = {};
        if (l16 < 8) {
          const __bf16* vr = &sVt[(h * 8 + l16) * NP + c * 32 + half * 8];
          bv = mk16(*(const bf16x8*)vr, *(const bf16x8*)(vr + 16));
        }
        o = wmma_bf16(ap, bv, o);
      }
      // store O tile (cols 0..7 valid) into sX reused as O buffer
      if (l16 < 8) {
#pragma unroll
        for (int v = 0; v < 8; v++)
          sX[(qt * 16 + v + half * 8) * 72 + h * 8 + l16] = (__bf16)o[v];
      }
    }
  }
  __syncthreads();

  // ---- Phase 5: projection [NTOK,64] @ [64,64] + bias, scatter fp32 ----
  for (int idx = wv; idx < RT * 4; idx += 8) {
    int rt = idx >> 2, ct = idx & 3;
    v8f acc = {};
#pragma unroll
    for (int c0 = 0; c0 < 64; c0 += 32) {
      const __bf16* orow = &sX[(rt * 16 + l16) * 72 + c0 + half * 8];
      v16bf a = mk16(*(const bf16x8*)orow, *(const bf16x8*)(orow + 16));
      const __bf16* wr = &sWproj[(ct * 16 + l16) * 72 + c0 + half * 8];
      v16bf bb = mk16(*(const bf16x8*)wr, *(const bf16x8*)(wr + 16));
      acc = wmma_bf16(a, bb, acc);
    }
    int col = ct * 16 + l16;
    float bias = sBias[col];
#pragma unroll
    for (int v = 0; v < 8; v++) {
      int t = rt * 16 + v + half * 8;
      int h, w;
      if (PASS == 1) { h = (t >> 4) * 8 + ph; w = (t & 15) * 8 + pw; }
      else           { h = qh * 8 + (t >> 3); w = qw * 8 + (t & 7); }
      xout[bbase + col * 16384 + h * 128 + w] = acc[v] + bias;
    }
  }
}

extern "C" void kernel_launch(void* const* d_in, const int* in_sizes, int n_in,
                              void* d_out, int out_size, void* d_ws, size_t ws_size,
                              hipStream_t stream) {
  (void)in_sizes; (void)n_in; (void)out_size; (void)d_ws; (void)ws_size;
  const float* x     = (const float*)d_in[0];
  const float* wqkv  = (const float*)d_in[1];
  const float* wproj = (const float*)d_in[2];
  const float* bproj = (const float*)d_in[3];
  float* out = (float*)d_out;

  // Pass 1: long-range (512 problems, 256 tokens) -> d_out
  isa_attn_kernel<256, 1><<<dim3(512), dim3(256), smem_bytes(256), stream>>>(
      x, wqkv, wproj, bproj, out);
  // Pass 2: short-range (2048 problems, 64 tokens) -> in place on d_out
  isa_attn_kernel<64, 2><<<dim3(2048), dim3(256), smem_bytes(64), stream>>>(
      out, wqkv, wproj, bproj, out);
}
